// LSTM_11038065951239
// MI455X (gfx1250) — matrix operand order repaired and checked by reading, not compiled, not run on hardware
//
#include <hip/hip_runtime.h>
#include <cstdint>
#include <cstddef>

// ---------------------------------------------------------------------------
// MI455X (gfx1250) persistent-LSTM design:
//  * wave32, WMMA V_WMMA_F32_16X16X32_F16 for all recurrent GEMMs
//  * Whh (and Wih for layer 1) converted to f16 once and kept resident in LDS
//    in WMMA B-fragment order (64 WGs x ~284KB LDS < 320KB/WGP)
//  * carry-h == ys[t] (matches reference scan), stored f16 in workspace and
//    re-read as WMMA A fragments next step
//  * per-step grid-wide sync via monotonic global atomic barrier
// ---------------------------------------------------------------------------

typedef __attribute__((ext_vector_type(16))) _Float16 v16h;
typedef __attribute__((ext_vector_type(8)))  float    v8f;
typedef __attribute__((ext_vector_type(4)))  unsigned u32x4;

namespace {
constexpr int T_STEPS = 1024;
constexpr int BATCH   = 64;
constexpr int HID     = 1024;
constexpr int IN_DIM  = 5;
constexpr int NWG     = 64;     // each WG owns 16 hidden units (64 gate rows)
constexpr int NTHR    = 256;    // 8 waves

// LDS layout (bytes)
constexpr int OFF_WHH   = 0;        // f16 WMMA B-fragments of Whh slice   (128KB)
constexpr int OFF_WIH   = 131072;   // f16 WMMA B-fragments of Wih slice   (128KB, layer1)
constexpr int OFF_GATES = 262144;   // f32 gate tiles [4][64][16]          (16KB)
constexpr int OFF_C     = 278528;   // f32 c state [64][16]                (4KB)
constexpr int OFF_H     = 282624;   // f32 h carry [64][16]                (4KB)
constexpr int OFF_X     = 286720;   // f32 x_t staging [64][8]             (2KB, layer0)
constexpr int OFF_W0    = 288768;   // f32 Wih0 slice [64][8]              (2KB, layer0)
constexpr int OFF_BIAS  = 290816;   // f32 bih+bhh slice [64]
constexpr int OFF_LENS  = 291072;   // i32 lens [64]
constexpr int SMEM_BYTES = 291328;  // < 320KB WGP LDS
} // namespace

__device__ __forceinline__ uint32_t pack2h(float a, float b) {
  union { _Float16 h[2]; uint32_t u; } x;
  x.h[0] = (_Float16)a; x.h[1] = (_Float16)b;
  return x.u;
}
__device__ __forceinline__ float sigf(float x) { return 1.f / (1.f + __expf(-x)); }

// Fill LDS with a (4 gates x 16 cols) x 1024-K weight slice, pre-swizzled into
// the 16-bit WMMA B-matrix (32x16) per-lane fragment layout:
//   dword r<4, lanes 0-15: K = 2r,2r+1 ; lanes 16-31: K = 8+2r,8+2r+1 ; dword r>=4: K += 8
__device__ void fill_frag(uint32_t* dst, const float* __restrict__ W, int j0) {
  for (int idx = threadIdx.x; idx < 4 * 32 * 32 * 8; idx += NTHR) {
    const int dw   = idx & 7;
    const int lane = (idx >> 3) & 31;
    const int kc   = (idx >> 8) & 31;
    const int g    = (idx >> 13) & 3;
    const int nl   = lane & 15;
    const int kl   = ((lane >> 4) << 3) + 2 * dw + (dw >= 4 ? 8 : 0);
    const size_t off = (size_t)(g * HID + j0 + nl) * HID + (size_t)(kc * 32 + kl);
    dst[idx] = pack2h(W[off], W[off + 1]);
  }
}

__global__ __launch_bounds__(NTHR) void lstm_layer_kernel(
    int layer,
    const float* __restrict__ data,     // layer0: (T,B,IN)
    const float* __restrict__ Wih,      // layer0: (4H,IN) ; layer1: (4H,H)
    const float* __restrict__ Whh,      // (4H,H)
    const float* __restrict__ bih,
    const float* __restrict__ bhh,
    const int*   __restrict__ lens,
    const _Float16* __restrict__ xsrc,  // layer1: ys of previous layer (T,B,H)
    _Float16* __restrict__ ys,          // this layer's carry/output (T,B,H)
    unsigned* __restrict__ bar)
{
  extern __shared__ char smem[];
  uint32_t* whhF  = (uint32_t*)(smem + OFF_WHH);
  uint32_t* wihF  = (uint32_t*)(smem + OFF_WIH);
  float*    sg    = (float*)(smem + OFF_GATES);
  float*    sc    = (float*)(smem + OFF_C);
  float*    shh   = (float*)(smem + OFF_H);
  float*    sx    = (float*)(smem + OFF_X);
  float*    sw0   = (float*)(smem + OFF_W0);
  float*    sbias = (float*)(smem + OFF_BIAS);
  int*      slens = (int*)(smem + OFF_LENS);

  const int tid = threadIdx.x;
  const int j0  = blockIdx.x * 16;   // this WG's first hidden unit

  // ---------------- prologue: weights -> LDS fragments, state init ----------
  fill_frag(whhF, Whh, j0);
  if (layer != 0) fill_frag(wihF, Wih, j0);
  for (int i = tid; i < BATCH * 16; i += NTHR) { sc[i] = 0.f; shh[i] = 0.f; }
  if (tid < 64) {
    const int g = tid >> 4, nl = tid & 15;
    const int col = g * HID + j0 + nl;
    sbias[tid] = bih[col] + bhh[col];
    slens[tid] = lens[tid];
    if (layer == 0) {
#pragma unroll
      for (int k = 0; k < IN_DIM; ++k) sw0[tid * 8 + k] = Wih[(size_t)col * IN_DIM + k];
    }
  }
  __syncthreads();

  const int wave = tid >> 5, lane = tid & 31;
  const int gI = wave & 3;           // which gate (i,f,g,o)
  const int mp = wave >> 2;          // batch half (0: b 0..31, 1: b 32..63)
  const int nl = lane & 15;
  const int laneHi = lane >> 4;

  for (int t = 0; t < T_STEPS; ++t) {
    v8f acc0 = {};                   // batches mp*32 + 0..15
    v8f acc1 = {};                   // batches mp*32 + 16..31

    auto gemm = [&](const _Float16* __restrict__ A, int trow,
                    const uint32_t* __restrict__ fragW) {
      // A fragment (16x32 f16): dwords 0-3 = K k0..k0+7, dwords 4-7 = k0+16..k0+23
      const size_t r0 = ((size_t)trow * BATCH + (size_t)(mp * 32 + nl)) * HID;
      const size_t r1 = r0 + (size_t)16 * HID;
      const u32x4* fb = (const u32x4*)fragW;
#pragma unroll 2
      for (int kc = 0; kc < 32; ++kc) {
        union { u32x4 q[2]; v16h h; } bf, a0, a1;
        const u32x4* bp = fb + (size_t)((gI * 32 + kc) * 32 + lane) * 2;
        bf.q[0] = bp[0]; bf.q[1] = bp[1];                  // 2x ds_load_b128
        const int k0 = kc * 32 + laneHi * 8;
        a0.q[0] = *(const u32x4*)(A + r0 + k0);            // global_load_b128
        a0.q[1] = *(const u32x4*)(A + r0 + k0 + 16);
        a1.q[0] = *(const u32x4*)(A + r1 + k0);
        a1.q[1] = *(const u32x4*)(A + r1 + k0 + 16);
        acc0 = __builtin_amdgcn_wmma_f32_16x16x32_f16(false, a0.h, false, bf.h,
                                                      (short)0, acc0, false, false);
        acc1 = __builtin_amdgcn_wmma_f32_16x16x32_f16(false, a1.h, false, bf.h,
                                                      (short)0, acc1, false, false);
      }
    };

    if (t > 0)      gemm(ys,   t - 1, whhF);   // h_{t-1} @ Whh^T   (h_0 == 0)
    if (layer != 0) gemm(xsrc, t,     wihF);   // x_t     @ Wih^T   (layer 1)

    // C-tile (16x16 f32) -> LDS gates: VGPR r, lanes 0-15: M=r ; 16-31: M=8+r
#pragma unroll
    for (int r = 0; r < 8; ++r) {
      const int m0 = mp * 32 + r + laneHi * 8;
      sg[(gI * 64 + m0) * 16 + nl]        = acc0[r];
      sg[(gI * 64 + m0 + 16) * 16 + nl]   = acc1[r];
    }
    __syncthreads();

    if (layer == 0 && tid < BATCH) {
#pragma unroll
      for (int k = 0; k < IN_DIM; ++k)
        sx[tid * 8 + k] = data[((size_t)t * BATCH + tid) * IN_DIM + k];
    }
    __syncthreads();

    // ---------------- gate nonlinearities + state update ---------------------
#pragma unroll
    for (int q = 0; q < 4; ++q) {
      const int p = q * NTHR + tid;       // 1024 (batch, hidden) pairs
      const int b = p & 63, jl = p >> 6;
      float gp[4];
#pragma unroll
      for (int gg = 0; gg < 4; ++gg)
        gp[gg] = sg[(gg * 64 + b) * 16 + jl] + sbias[gg * 16 + jl];
      if (layer == 0) {                   // tiny K=5 input projection
#pragma unroll
        for (int gg = 0; gg < 4; ++gg) {
          float d = 0.f;
#pragma unroll
          for (int k = 0; k < IN_DIM; ++k) d += sx[b * 8 + k] * sw0[(gg * 16 + jl) * 8 + k];
          gp[gg] += d;
        }
      }
      const float ig = sigf(gp[0]), fg = sigf(gp[1]);
      const float gt = tanhf(gp[2]), og = sigf(gp[3]);
      const int ci = b * 16 + jl;
      const float cold = sc[ci], hold = shh[ci];
      const float cnew = fg * cold + ig * gt;
      const float hnew = og * tanhf(cnew);
      const bool  v  = t < slens[b];
      const float cc = v ? cnew : cold;
      const float hh = v ? hnew : hold;   // carry == stored ys[t] (ref semantics)
      sc[ci] = cc; shh[ci] = hh;
      ys[((size_t)t * BATCH + b) * HID + j0 + jl] = (_Float16)hh;
    }

    // ---------------- grid-wide step barrier (monotonic counter) -------------
    __threadfence();
    __syncthreads();
    if (tid == 0) {
      atomicAdd(bar, 1u);
      const unsigned target = (unsigned)gridDim.x * (unsigned)(t + 1);
      while (atomicAdd(bar, 0u) < target) __builtin_amdgcn_s_sleep(8);
    }
    __syncthreads();
    __threadfence();
  }
}

// Output projection (5-wide) + softmax over channels 2..4; wave-per-row.
__global__ __launch_bounds__(256) void proj_kernel(
    const _Float16* __restrict__ ysL, const int* __restrict__ lens,
    const float* __restrict__ Wout, const float* __restrict__ bout,
    float* __restrict__ out)
{
  __shared__ float sW[IN_DIM * HID];
  for (int i = threadIdx.x; i < IN_DIM * HID; i += 256) sW[i] = Wout[i];
  __syncthreads();
  const int wave = threadIdx.x >> 5, lane = threadIdx.x & 31;
  const int row = blockIdx.x * 8 + wave;     // row = t*B + b
  const int t = row >> 6, b = row & 63;
  float s0 = 0.f, s1 = 0.f, s2 = 0.f, s3 = 0.f, s4 = 0.f;
  if (t < lens[b]) {                          // invalid rows -> ys masked to 0
    const _Float16* h = ysL + (size_t)row * HID;
    for (int k = lane; k < HID; k += 32) {
      const float x = (float)h[k];
      s0 += x * sW[0 * HID + k]; s1 += x * sW[1 * HID + k];
      s2 += x * sW[2 * HID + k]; s3 += x * sW[3 * HID + k];
      s4 += x * sW[4 * HID + k];
    }
  }
#pragma unroll
  for (int m = 16; m >= 1; m >>= 1) {
    s0 += __shfl_xor(s0, m, 32); s1 += __shfl_xor(s1, m, 32);
    s2 += __shfl_xor(s2, m, 32); s3 += __shfl_xor(s3, m, 32);
    s4 += __shfl_xor(s4, m, 32);
  }
  if (lane == 0) {
    const float l0 = s0 + bout[0], l1 = s1 + bout[1];
    const float l2 = s2 + bout[2], l3 = s3 + bout[3], l4 = s4 + bout[4];
    const float mx = fmaxf(l2, fmaxf(l3, l4));
    const float e2 = __expf(l2 - mx), e3 = __expf(l3 - mx), e4 = __expf(l4 - mx);
    const float inv = 1.f / (e2 + e3 + e4);
    float* o = out + (size_t)row * IN_DIM;
    o[0] = l0; o[1] = l1; o[2] = e2 * inv; o[3] = e3 * inv; o[4] = e4 * inv;
  }
}

extern "C" void kernel_launch(void* const* d_in, const int* in_sizes, int n_in,
                              void* d_out, int out_size, void* d_ws, size_t ws_size,
                              hipStream_t stream)
{
  (void)in_sizes; (void)n_in; (void)out_size; (void)ws_size;
  const float* data = (const float*)d_in[0];
  const int*   lens = (const int*)d_in[1];
  const float* Wih0 = (const float*)d_in[2];
  const float* Whh0 = (const float*)d_in[3];
  const float* bih0 = (const float*)d_in[4];
  const float* bhh0 = (const float*)d_in[5];
  const float* Wih1 = (const float*)d_in[6];   // (L-1)=1 slice
  const float* Whh1 = (const float*)d_in[7];
  const float* bih1 = (const float*)d_in[8];
  const float* bhh1 = (const float*)d_in[9];
  const float* Wout = (const float*)d_in[10];
  const float* bout = (const float*)d_in[11];
  float* out = (float*)d_out;

  // Workspace: [0,512) barrier L0, [512,1024) barrier L1,
  //            then ys0 (T*B*H f16 = 128MB), ys1 (128MB).
  char* ws = (char*)d_ws;
  unsigned* bar0 = (unsigned*)(ws);
  unsigned* bar1 = (unsigned*)(ws + 512);
  _Float16* ys0 = (_Float16*)(ws + 1024);
  _Float16* ys1 = ys0 + (size_t)T_STEPS * BATCH * HID;

  (void)hipMemsetAsync(ws, 0, 1024, stream);   // reset barrier counters
  (void)hipFuncSetAttribute((const void*)lstm_layer_kernel,
                            hipFuncAttributeMaxDynamicSharedMemorySize,
                            (int)SMEM_BYTES);

  lstm_layer_kernel<<<NWG, NTHR, SMEM_BYTES, stream>>>(
      0, data, Wih0, Whh0, bih0, bhh0, lens, nullptr, ys0, bar0);
  lstm_layer_kernel<<<NWG, NTHR, SMEM_BYTES, stream>>>(
      1, nullptr, Wih1, Whh1, bih1, bhh1, lens, ys0, ys1, bar1);
  proj_kernel<<<(T_STEPS * BATCH) / 8, 256, 0, stream>>>(ys1, lens, Wout, bout, out);
}
